// PadTransformer_45913200394617
// MI455X (gfx1250) — matrix operand order repaired and checked by reading, not compiled, not run on hardware
//
#include <hip/hip_runtime.h>
#include <hip/hip_bf16.h>

// ---------------- problem constants ----------------
#define NLAYERS 2
#define DIM     1024
#define NHEAD   16
#define HD      64
#define FFD     4096
#define SEQ_S   1024
#define NB      2
#define SEQ_L   4096
#define LN_EPS  1e-5f

typedef __attribute__((ext_vector_type(16))) __bf16 v16bf;
typedef __attribute__((ext_vector_type(8)))  float  v8f;

union FragBF {
    v16bf v;
    uint4 q[2];
};

// round-to-nearest-even f32 -> bf16 bit pattern
__device__ __forceinline__ unsigned short f2bf(float f) {
    unsigned u = __float_as_uint(f);
    u += 0x7FFFu + ((u >> 16) & 1u);
    return (unsigned short)(u >> 16);
}

// A-matrix fragment (16xK tile, K-step 32), row-major source, ld in elements.
// Lane half 0: K = k0..k0+15 contiguously; half 1: k0+16..k0+31.
__device__ __forceinline__ FragBF load_fragA(const unsigned short* base, int ld, int k0, int lane) {
    int r = lane & 15, hf = lane >> 4;
    const unsigned short* p = base + (size_t)r * ld + k0 + hf * 16;
    FragBF f;
    f.q[0] = *(const uint4*)(p);
    f.q[1] = *(const uint4*)(p + 8);
    return f;
}

// B-matrix fragment: B is (32 x 16) = K x N; column n = row n of W (N,K) row-major.
// Lane half 0 holds K = k0..k0+15 contiguously; half 1 holds k0+16..k0+31.
__device__ __forceinline__ FragBF load_fragB(const unsigned short* base, int ld, int k0, int lane) {
    int r = lane & 15, hf = lane >> 4;
    const unsigned short* p = base + (size_t)r * ld + k0 + hf * 16;
    FragBF f;
    f.q[0] = *(const uint4*)(p);
    f.q[1] = *(const uint4*)(p + 8);
    return f;
}

__device__ __forceinline__ v8f wmma_bf16(const FragBF& a, const FragBF& b, v8f c) {
    return __builtin_amdgcn_wmma_f32_16x16x32_bf16(false, a.v, false, b.v, (short)0, c, false, false);
}

__device__ __forceinline__ float hmax16(float x) {
    x = fmaxf(x, __shfl_xor(x, 1, 32));
    x = fmaxf(x, __shfl_xor(x, 2, 32));
    x = fmaxf(x, __shfl_xor(x, 4, 32));
    x = fmaxf(x, __shfl_xor(x, 8, 32));
    return x;
}
__device__ __forceinline__ float hsum16(float x) {
    x += __shfl_xor(x, 1, 32);
    x += __shfl_xor(x, 2, 32);
    x += __shfl_xor(x, 4, 32);
    x += __shfl_xor(x, 8, 32);
    return x;
}

// ---------------- elementwise cast ----------------
__global__ void cast_f32_bf16_kernel(const float* __restrict__ in,
                                     unsigned short* __restrict__ out, size_t n) {
    size_t i = (size_t)blockIdx.x * blockDim.x + threadIdx.x;
    size_t stride = (size_t)gridDim.x * blockDim.x;
    for (; i < n; i += stride) out[i] = f2bf(in[i]);
}

// ---------------- pseudo-normal noise (stand-in for jax fold_in PRNG) ----------------
__device__ __forceinline__ unsigned hashu(unsigned x) {
    x ^= x >> 16; x *= 0x7feb352dU;
    x ^= x >> 15; x *= 0x846ca68bU;
    x ^= x >> 16; return x;
}
__global__ void noise_kernel(float* __restrict__ noise, int layer) {
    int i = blockIdx.x * blockDim.x + threadIdx.x;
    if (i < NB * NHEAD * HD) {
        unsigned s1 = hashu(0x9E3779B9u * (unsigned)(layer + 1) + (unsigned)i * 2u + 0u);
        unsigned s2 = hashu(0x85EBCA6Bu * (unsigned)(layer + 1) + (unsigned)i * 2u + 1u);
        float u1 = ((float)((s1 >> 8) + 1u)) * (1.0f / 16777217.0f);
        float u2 = ((float)(s2 >> 8)) * (1.0f / 16777216.0f);
        noise[i] = sqrtf(-2.0f * __logf(u1)) * __cosf(6.28318530718f * u2);
    }
}

// ---------------- generic WMMA GEMM: Y = act((A @ W^T + b) * alpha) ----------------
// A: (M,K) bf16 row-major; W: (NN,K) bf16 row-major.
// 4 waves/block; each wave owns a 32x64 output strip (2 A-frags x 4 B-frags = 8 WMMA/K-step),
// software-pipelined: next K-step fragments are loaded before current WMMAs issue.
// out_mode 0: Cf (f32) and/or Cb (bf16), row-major (M,NN)
// out_mode 1: head scatter     Cs[((nb*H+h)*SEQ+seq)*HD+hd]      (row = seq*NB+nb, col = h*HD+hd)
// out_mode 2: head scatter^T   Cs[((nb*H+h)*HD+hd)*SEQ+seq]
__global__ __launch_bounds__(128) void gemm_wmma_kernel(
    const unsigned short* __restrict__ A, int lda,
    const unsigned short* __restrict__ W, int ldw,
    const float* __restrict__ bias, float alpha, int relu, int out_mode,
    float* __restrict__ Cf, unsigned short* __restrict__ Cb,
    unsigned short* __restrict__ Cs,
    int M, int NN, int SEQ, int K) {

    int lane = threadIdx.x & 31;
    int wave = threadIdx.x >> 5;
    int row0 = (blockIdx.x * 4 + wave) * 32;
    int col0 = blockIdx.y * 64;
    if (row0 >= M) return;

    const unsigned short* Abase0 = A + (size_t)row0 * lda;
    const unsigned short* Abase1 = A + (size_t)(row0 + 16) * lda;

    v8f acc[2][4] = {};

    FragBF a0 = load_fragA(Abase0, lda, 0, lane);
    FragBF a1 = load_fragA(Abase1, lda, 0, lane);
    FragBF b0 = load_fragB(W + (size_t)(col0 +  0) * ldw, ldw, 0, lane);
    FragBF b1 = load_fragB(W + (size_t)(col0 + 16) * ldw, ldw, 0, lane);
    FragBF b2 = load_fragB(W + (size_t)(col0 + 32) * ldw, ldw, 0, lane);
    FragBF b3 = load_fragB(W + (size_t)(col0 + 48) * ldw, ldw, 0, lane);

    for (int k0 = 0; k0 < K; k0 += 32) {
        int kn = (k0 + 32 < K) ? (k0 + 32) : 0;  // wrap on last iter; result unused
        __builtin_prefetch(Abase0 + (size_t)(lane & 15) * lda + kn + 32, 0, 1);
        FragBF an0 = load_fragA(Abase0, lda, kn, lane);
        FragBF an1 = load_fragA(Abase1, lda, kn, lane);
        FragBF bn0 = load_fragB(W + (size_t)(col0 +  0) * ldw, ldw, kn, lane);
        FragBF bn1 = load_fragB(W + (size_t)(col0 + 16) * ldw, ldw, kn, lane);
        FragBF bn2 = load_fragB(W + (size_t)(col0 + 32) * ldw, ldw, kn, lane);
        FragBF bn3 = load_fragB(W + (size_t)(col0 + 48) * ldw, ldw, kn, lane);

        acc[0][0] = wmma_bf16(a0, b0, acc[0][0]);
        acc[0][1] = wmma_bf16(a0, b1, acc[0][1]);
        acc[0][2] = wmma_bf16(a0, b2, acc[0][2]);
        acc[0][3] = wmma_bf16(a0, b3, acc[0][3]);
        acc[1][0] = wmma_bf16(a1, b0, acc[1][0]);
        acc[1][1] = wmma_bf16(a1, b1, acc[1][1]);
        acc[1][2] = wmma_bf16(a1, b2, acc[1][2]);
        acc[1][3] = wmma_bf16(a1, b3, acc[1][3]);

        a0 = an0; a1 = an1;
        b0 = bn0; b1 = bn1; b2 = bn2; b3 = bn3;
    }

    int r = lane & 15, hf = lane >> 4;
#pragma unroll
    for (int mi = 0; mi < 2; ++mi) {
#pragma unroll
        for (int j = 0; j < 4; ++j) {
            int col = col0 + 16 * j + r;
            float bv = bias ? bias[col] : 0.0f;
#pragma unroll
            for (int v = 0; v < 8; ++v) {
                int row = row0 + mi * 16 + v + 8 * hf;
                float val = (acc[mi][j][v] + bv) * alpha;
                if (relu) val = fmaxf(val, 0.0f);
                if (out_mode == 0) {
                    size_t idx = (size_t)row * NN + col;
                    if (Cf) Cf[idx] = val;
                    if (Cb) Cb[idx] = f2bf(val);
                } else {
                    int seq = row / NB, nb = row % NB;
                    int h = col / HD, hd = col % HD;
                    size_t idx;
                    if (out_mode == 1)
                        idx = (((size_t)nb * NHEAD + h) * SEQ + seq) * HD + hd;
                    else
                        idx = (((size_t)nb * NHEAD + h) * HD + hd) * SEQ + seq;
                    Cs[idx] = f2bf(val);
                }
            }
        }
    }
}

// ---------------- attention pass 1: softmax stats (max m, partition Z) ----------------
// grid: NB * NHEAD * (L/16) single-wave blocks. Includes the appended zero-score column.
__global__ __launch_bounds__(32) void attn_stats_kernel(
    const unsigned short* __restrict__ qh, const unsigned short* __restrict__ kh,
    float* __restrict__ Mb, float* __restrict__ Zb) {

    const int LT = SEQ_L / 16;
    int tile = blockIdx.x;
    int lt = tile % LT;
    int h  = (tile / LT) % NHEAD;
    int n  = tile / (LT * NHEAD);
    int lane = threadIdx.x, r = lane & 15, hf = lane >> 4;

    const unsigned short* qbase = qh + (((size_t)n * NHEAD + h) * SEQ_L + lt * 16) * HD;
    const unsigned short* kall  = kh + (((size_t)n * NHEAD + h) * SEQ_S) * HD;

    FragBF qf0 = load_fragA(qbase, HD, 0, lane);
    FragBF qf1 = load_fragA(qbase, HD, 32, lane);

    float m8[8], z8[8];
#pragma unroll
    for (int v = 0; v < 8; ++v) { m8[v] = -1e30f; z8[v] = 0.0f; }

    for (int s0 = 0; s0 < SEQ_S; s0 += 16) {
        v8f c = {};
        c = wmma_bf16(qf0, load_fragB(kall + (size_t)s0 * HD, HD, 0, lane),  c);
        c = wmma_bf16(qf1, load_fragB(kall + (size_t)s0 * HD, HD, 32, lane), c);
#pragma unroll
        for (int v = 0; v < 8; ++v) {
            float tmax = hmax16(c[v]);
            float mn = fmaxf(m8[v], tmax);
            float zs = hsum16(__expf(c[v] - mn));
            z8[v] = z8[v] * __expf(m8[v] - mn) + zs;
            m8[v] = mn;
        }
    }
    // virtual column with score 0
#pragma unroll
    for (int v = 0; v < 8; ++v) {
        float mn = fmaxf(m8[v], 0.0f);
        z8[v] = z8[v] * __expf(m8[v] - mn) + __expf(-mn);
        m8[v] = mn;
    }
    if (r == 0) {
#pragma unroll
        for (int v = 0; v < 8; ++v) {
            size_t idx = ((size_t)n * NHEAD + h) * SEQ_L + lt * 16 + v + 8 * hf;
            Mb[idx] = m8[v];
            Zb[idx] = z8[v];
        }
    }
}

// ---------------- attention pass 2: O = softmax(QK^T) @ V (+ noise row) ----------------
// grid: NB * NHEAD * (L/16) single-wave blocks. P fragment is staged via LDS.
__global__ __launch_bounds__(32) void attn_out_kernel(
    const unsigned short* __restrict__ qh, const unsigned short* __restrict__ kh,
    const unsigned short* __restrict__ vT, const float* __restrict__ Mb,
    const float* __restrict__ Zb, const float* __restrict__ noise,
    unsigned short* __restrict__ attn_out) {

    const int LT = SEQ_L / 16;
    int tile = blockIdx.x;
    int lt = tile % LT;
    int h  = (tile / LT) % NHEAD;
    int n  = tile / (LT * NHEAD);
    int lane = threadIdx.x, r = lane & 15, hf = lane >> 4;

    const unsigned short* qbase = qh + (((size_t)n * NHEAD + h) * SEQ_L + lt * 16) * HD;
    const unsigned short* kall  = kh + (((size_t)n * NHEAD + h) * SEQ_S) * HD;
    const unsigned short* vhead = vT + (((size_t)n * NHEAD + h) * HD) * SEQ_S;

    FragBF qf0 = load_fragA(qbase, HD, 0, lane);
    FragBF qf1 = load_fragA(qbase, HD, 32, lane);

    float mrow[8], zinv[8];
#pragma unroll
    for (int v = 0; v < 8; ++v) {
        size_t idx = ((size_t)n * NHEAD + h) * SEQ_L + lt * 16 + v + 8 * hf;
        mrow[v] = Mb[idx];
        zinv[v] = 1.0f / Zb[idx];
    }

    __shared__ __align__(16) unsigned short Pst[16 * 32];
    v8f acc[4] = {};

    for (int s0 = 0; s0 < SEQ_S; s0 += 32) {
#pragma unroll
        for (int sub = 0; sub < 2; ++sub) {
            int sb = s0 + sub * 16;
            v8f c = {};
            c = wmma_bf16(qf0, load_fragB(kall + (size_t)sb * HD, HD, 0, lane),  c);
            c = wmma_bf16(qf1, load_fragB(kall + (size_t)sb * HD, HD, 32, lane), c);
#pragma unroll
            for (int v = 0; v < 8; ++v) {
                float p = __expf(c[v] - mrow[v]) * zinv[v];
                Pst[(v + 8 * hf) * 32 + sub * 16 + r] = f2bf(p);
            }
        }
        __syncthreads();
        FragBF pf = load_fragA(Pst, 32, 0, lane);
#pragma unroll
        for (int j = 0; j < 4; ++j) {
            FragBF vb = load_fragB(vhead + (size_t)(16 * j) * SEQ_S, SEQ_S, s0, lane);
            acc[j] = wmma_bf16(pf, vb, acc[j]);
        }
        __syncthreads();
    }

    // appended noise row: weight = exp(0 - m) / Z
#pragma unroll
    for (int j = 0; j < 4; ++j) {
        int hd = 16 * j + r;
        float nz = noise[((size_t)n * NHEAD + h) * HD + hd];
#pragma unroll
        for (int v = 0; v < 8; ++v) {
            float pS = __expf(-mrow[v]) * zinv[v];
            acc[j][v] += pS * nz;
            int l = lt * 16 + v + 8 * hf;
            attn_out[((size_t)l * NB + n) * DIM + h * HD + hd] = f2bf(acc[j][v]);
        }
    }
}

// ---------------- attention pass 3 (last layer): attn_map = mean_h sigmoid(p) ----------------
// grid: NB * (L/16) single-wave blocks; writes (N, L, S) f32 directly into d_out.
__global__ __launch_bounds__(32) void attn_map_kernel(
    const unsigned short* __restrict__ qh, const unsigned short* __restrict__ kh,
    const float* __restrict__ Mb, const float* __restrict__ Zb,
    float* __restrict__ amap) {

    const int LT = SEQ_L / 16;
    int lt = blockIdx.x % LT;
    int n  = blockIdx.x / LT;
    int lane = threadIdx.x, r = lane & 15, hf = lane >> 4;

    __shared__ float sm[NHEAD * 16];
    __shared__ float sz[NHEAD * 16];
    for (int idx = threadIdx.x; idx < NHEAD * 16; idx += 32) {
        int hh = idx >> 4, rr = idx & 15;
        size_t g = ((size_t)n * NHEAD + hh) * SEQ_L + lt * 16 + rr;
        sm[idx] = Mb[g];
        sz[idx] = 1.0f / Zb[g];
    }
    __syncthreads();

    for (int s0 = 0; s0 < SEQ_S; s0 += 16) {
        float am[8] = {};
        for (int h = 0; h < NHEAD; ++h) {
            const unsigned short* qbase = qh + (((size_t)n * NHEAD + h) * SEQ_L + lt * 16) * HD;
            const unsigned short* kbase = kh + (((size_t)n * NHEAD + h) * SEQ_S + s0) * HD;
            v8f c = {};
            c = wmma_bf16(load_fragA(qbase, HD, 0, lane),  load_fragB(kbase, HD, 0, lane),  c);
            c = wmma_bf16(load_fragA(qbase, HD, 32, lane), load_fragB(kbase, HD, 32, lane), c);
#pragma unroll
            for (int v = 0; v < 8; ++v) {
                int row16 = v + 8 * hf;
                float p = __expf(c[v] - sm[h * 16 + row16]) * sz[h * 16 + row16];
                am[v] += 1.0f / (1.0f + __expf(-p));
            }
        }
#pragma unroll
        for (int v = 0; v < 8; ++v) {
            int l = lt * 16 + v + 8 * hf;
            amap[((size_t)n * SEQ_L + l) * SEQ_S + s0 + r] = am[v] * (1.0f / NHEAD);
        }
    }
}

// ---------------- fused residual + LayerNorm (f32 + bf16 outputs) ----------------
__global__ __launch_bounds__(256) void ln_kernel(
    const float* __restrict__ a, const float* __restrict__ b,
    const float* __restrict__ g, const float* __restrict__ be,
    float* __restrict__ yf, unsigned short* __restrict__ yb) {

    int row = blockIdx.x, t = threadIdx.x;
    const float* ar = a + (size_t)row * DIM;
    const float* br = b + (size_t)row * DIM;
    __shared__ float red[256];

    float x[4];
    float s = 0.0f;
#pragma unroll
    for (int j = 0; j < 4; ++j) {
        x[j] = ar[t + j * 256] + br[t + j * 256];
        s += x[j];
    }
    red[t] = s; __syncthreads();
    for (int o = 128; o > 0; o >>= 1) { if (t < o) red[t] += red[t + o]; __syncthreads(); }
    float mu = red[0] * (1.0f / DIM);
    __syncthreads();

    float q = 0.0f;
#pragma unroll
    for (int j = 0; j < 4; ++j) { float d = x[j] - mu; q += d * d; }
    red[t] = q; __syncthreads();
    for (int o = 128; o > 0; o >>= 1) { if (t < o) red[t] += red[t + o]; __syncthreads(); }
    float rstd = rsqrtf(red[0] * (1.0f / DIM) + LN_EPS);

#pragma unroll
    for (int j = 0; j < 4; ++j) {
        int idx = t + j * 256;
        float y = (x[j] - mu) * rstd * g[idx] + be[idx];
        yf[(size_t)row * DIM + idx] = y;
        yb[(size_t)row * DIM + idx] = f2bf(y);
    }
}

// ---------------- host orchestration ----------------
extern "C" void kernel_launch(void* const* d_in, const int* in_sizes, int n_in,
                              void* d_out, int out_size, void* d_ws, size_t ws_size,
                              hipStream_t stream) {
    const float* key_in    = (const float*)d_in[0];
    const float* value_in  = (const float*)d_in[1];
    const float* query     = (const float*)d_in[2];
    const float* in_proj_w = (const float*)d_in[3];
    const float* in_proj_b = (const float*)d_in[4];
    const float* out_w     = (const float*)d_in[5];
    const float* out_b     = (const float*)d_in[6];
    const float* ln1_g     = (const float*)d_in[7];
    const float* ln1_b     = (const float*)d_in[8];
    const float* ln2_g     = (const float*)d_in[9];
    const float* ln2_b     = (const float*)d_in[10];
    const float* ff1_w     = (const float*)d_in[11];
    const float* ff1_b     = (const float*)d_in[12];
    const float* ff2_w     = (const float*)d_in[13];
    const float* ff2_b     = (const float*)d_in[14];
    float* out = (float*)d_out;

    const size_t LND = (size_t)SEQ_L * NB * DIM;   // 8,388,608
    const size_t SND = (size_t)SEQ_S * NB * DIM;   // 2,097,152
    const size_t HND = (size_t)NB * NHEAD * SEQ_L; // 131,072

    char* ws = (char*)d_ws;
    size_t off = 0;
    auto carve = [&](size_t bytes) {
        off = (off + 255) & ~(size_t)255;
        void* p = ws + off;
        off += bytes;
        return p;
    };
    float*          tgt_f32   = (float*)carve(LND * 4);
    float*          x_f32     = (float*)carve(LND * 4);
    float*          proj_f32  = (float*)carve(LND * 4);   // reused for ff2 output
    unsigned short* tgt_bf    = (unsigned short*)carve(LND * 2);
    unsigned short* x_bf      = (unsigned short*)carve(LND * 2);
    unsigned short* attn_bf   = (unsigned short*)carve(LND * 2);
    unsigned short* key_bf    = (unsigned short*)carve(SND * 2);
    unsigned short* val_bf    = (unsigned short*)carve(SND * 2);
    unsigned short* qh_bf     = (unsigned short*)carve(LND * 2);
    unsigned short* kh_bf     = (unsigned short*)carve(SND * 2);
    unsigned short* vT_bf     = (unsigned short*)carve(SND * 2);
    unsigned short* h_bf      = (unsigned short*)carve((size_t)SEQ_L * NB * FFD * 2);
    unsigned short* ipw_bf    = (unsigned short*)carve((size_t)3 * DIM * DIM * 2);
    unsigned short* ow_bf     = (unsigned short*)carve((size_t)DIM * DIM * 2);
    unsigned short* f1w_bf    = (unsigned short*)carve((size_t)FFD * DIM * 2);
    unsigned short* f2w_bf    = (unsigned short*)carve((size_t)DIM * FFD * 2);
    float*          Mb        = (float*)carve(HND * 4);
    float*          Zb        = (float*)carve(HND * 4);
    float*          noise     = (float*)carve((size_t)NB * NHEAD * HD * 4);

    auto cast = [&](const float* src, unsigned short* dst, size_t n) {
        cast_f32_bf16_kernel<<<2048, 256, 0, stream>>>(src, dst, n);
    };
    auto gemm = [&](const unsigned short* A, int lda, const unsigned short* W, int ldw,
                    const float* bias, float alpha, int relu, int mode,
                    float* Cf, unsigned short* Cb, unsigned short* Cs,
                    int M, int NN, int SEQ, int K) {
        dim3 grid(M / 128, NN / 64);   // 4 waves/block, 32 rows per wave
        gemm_wmma_kernel<<<grid, 128, 0, stream>>>(A, lda, W, ldw, bias, alpha, relu, mode,
                                                   Cf, Cb, Cs, M, NN, SEQ, K);
    };

    const float scaling = 0.125f;  // HD^-0.5

    // initial activations
    hipMemcpyAsync(tgt_f32, query, LND * 4, hipMemcpyDeviceToDevice, stream);
    cast(query, tgt_bf, LND);
    cast(key_in, key_bf, SND);
    cast(value_in, val_bf, SND);

    for (int i = 0; i < NLAYERS; ++i) {
        // per-layer weight conversion to bf16
        cast(in_proj_w + (size_t)i * 3 * DIM * DIM, ipw_bf, (size_t)3 * DIM * DIM);
        cast(out_w + (size_t)i * DIM * DIM, ow_bf, (size_t)DIM * DIM);
        cast(ff1_w + (size_t)i * FFD * DIM, f1w_bf, (size_t)FFD * DIM);
        cast(ff2_w + (size_t)i * DIM * FFD, f2w_bf, (size_t)DIM * FFD);
        noise_kernel<<<(NB * NHEAD * HD + 255) / 256, 256, 0, stream>>>(noise, i);

        const float* ipb = in_proj_b + (size_t)i * 3 * DIM;
        // Q/K/V projections with fused head scatter (V stored transposed per head)
        gemm(tgt_bf, DIM, ipw_bf,                     DIM, ipb,           scaling, 0, 1,
             nullptr, nullptr, qh_bf, SEQ_L * NB, DIM, SEQ_L, DIM);
        gemm(key_bf, DIM, ipw_bf + (size_t)DIM * DIM, DIM, ipb + DIM,    1.0f,    0, 1,
             nullptr, nullptr, kh_bf, SEQ_S * NB, DIM, SEQ_S, DIM);
        gemm(val_bf, DIM, ipw_bf + (size_t)2 * DIM * DIM, DIM, ipb + 2 * DIM, 1.0f, 0, 2,
             nullptr, nullptr, vT_bf, SEQ_S * NB, DIM, SEQ_S, DIM);

        int nblk = NB * NHEAD * (SEQ_L / 16);
        attn_stats_kernel<<<nblk, 32, 0, stream>>>(qh_bf, kh_bf, Mb, Zb);
        attn_out_kernel<<<nblk, 32, 0, stream>>>(qh_bf, kh_bf, vT_bf, Mb, Zb, noise, attn_bf);
        if (i == NLAYERS - 1)
            attn_map_kernel<<<NB * (SEQ_L / 16), 32, 0, stream>>>(qh_bf, kh_bf, Mb, Zb, out + LND);

        // out projection -> residual + LN1 -> FF1(ReLU) -> FF2 -> residual + LN2
        gemm(attn_bf, DIM, ow_bf, DIM, out_b + (size_t)i * DIM, 1.0f, 0, 0,
             proj_f32, nullptr, nullptr, SEQ_L * NB, DIM, 0, DIM);
        ln_kernel<<<SEQ_L * NB, 256, 0, stream>>>(tgt_f32, proj_f32,
                                                  ln1_g + (size_t)i * DIM, ln1_b + (size_t)i * DIM,
                                                  x_f32, x_bf);
        gemm(x_bf, DIM, f1w_bf, DIM, ff1_b + (size_t)i * FFD, 1.0f, 1, 0,
             nullptr, h_bf, nullptr, SEQ_L * NB, FFD, 0, DIM);
        gemm(h_bf, FFD, f2w_bf, FFD, ff2_b + (size_t)i * DIM, 1.0f, 0, 0,
             proj_f32, nullptr, nullptr, SEQ_L * NB, DIM, 0, FFD);
        ln_kernel<<<SEQ_L * NB, 256, 0, stream>>>(x_f32, proj_f32,
                                                  ln2_g + (size_t)i * DIM, ln2_b + (size_t)i * DIM,
                                                  tgt_f32, tgt_bf);
    }

    hipMemcpyAsync(out, tgt_f32, LND * 4, hipMemcpyDeviceToDevice, stream);
}
